// GeminiFusionBackbone_82506321756440
// MI455X (gfx1250) — compile-verified, matrix operand order checked by reading, not run on hardware
//
#include <hip/hip_runtime.h>
#include <hip/hip_bf16.h>
#include <cstdint>

#define C_DIM 512
#define BATCH 8
#define SEQ   1024
#define BN    (BATCH * SEQ)   // 8192 rows
#define HEADS 8
#define HD    64

typedef __bf16 bf16_t;
typedef __attribute__((ext_vector_type(16))) __bf16 v16bf;
typedef __attribute__((ext_vector_type(8)))  __bf16 v8bf;
typedef __attribute__((ext_vector_type(8)))  float  v8f;
typedef __attribute__((ext_vector_type(4)))  unsigned int v4u;
typedef __attribute__((ext_vector_type(8)))  int v8i;
typedef __attribute__((ext_vector_type(4)))  int v4i;

// ---------------------------------------------------------------------------
// WMMA helpers (CDNA5 wave32, 16x16x32 bf16 -> f32)
// ---------------------------------------------------------------------------
__device__ __forceinline__ v8f wmma_bf16(v16bf a, v16bf b, v8f c) {
  return __builtin_amdgcn_wmma_f32_16x16x32_bf16(
      /*neg_a=*/false, a, /*neg_b=*/false, b,
      /*c_mod=*/(short)0, c, /*reuse_a=*/false, /*reuse_b=*/false);
}

// Load a 16x32 bf16 fragment in the documented A/B VGPR layout from a
// row-major matrix (rows = fragment "rows"/"cols", contiguous K).
// lane<16: K = k0+[0..7] and k0+[16..23]; lane>=16: K = k0+[8..15], k0+[24..31].
__device__ __forceinline__ v16bf load_frag_rm(const bf16_t* base, int row0,
                                              int ld, int k0) {
  int lane = threadIdx.x & 31;
  const bf16_t* p = base + (size_t)(row0 + (lane & 15)) * (size_t)ld +
                    (size_t)(k0 + ((lane >> 4) << 3));
  v8bf lo = *(const v8bf*)p;         // 16B load
  v8bf hi = *(const v8bf*)(p + 16);  // 16B load
  v16bf f;
#pragma unroll
  for (int i = 0; i < 8; ++i) { f[i] = lo[i]; f[8 + i] = hi[i]; }
  return f;
}

__device__ __forceinline__ float gelu_exact(float x) {
  return 0.5f * x * (1.0f + erff(x * 0.70710678118654752f));
}

// ---------------------------------------------------------------------------
// Tensor Data Mover: 2D tile load (64 halfs x 32 rows, row stride 2*C_DIM
// halfs) from global into contiguous LDS. D# per CDNA5 ISA 8.3/8.4.
// This toolchain exposes the 6-arg builtin:
//   (v4u g0, v8i g1, v4i g2, v4i g3, v8i g4, i32 cpol)
// ---------------------------------------------------------------------------
__device__ __forceinline__ void tdm_load_k_tile(const bf16_t* gsrc,
                                                bf16_t* lds_dst) {
  uint32_t lds_off = (uint32_t)(uintptr_t)lds_dst;  // addr[31:0] == LDS byte offset
  uint64_t ga = (uint64_t)(uintptr_t)gsrc;
  v4u g0;
  g0[0] = 1u;                                  // count=1, user descriptor
  g0[1] = lds_off;                             // lds_addr (bytes)
  g0[2] = (uint32_t)ga;                        // global_addr[31:0]
  g0[3] = (uint32_t)(ga >> 32) | (2u << 30);   // global_addr[56:32] | type=2
  v8i g1;
  g1[0] = 1 << 16;                             // data_size = 2B
  g1[1] = 64 << 16;                            // tensor_dim0 = 64
  g1[2] = 32 << 16;                            // tensor_dim1 = 32
  g1[3] = 64 << 16;                            // tile_dim0 = 64
  g1[4] = 32;                                  // tile_dim1 = 32 (tile_dim2 = 0)
  g1[5] = 2 * C_DIM;                           // tensor_dim0_stride = 1024
  g1[6] = 0;
  g1[7] = 0;
  v4i gz4 = {0, 0, 0, 0};
  v8i gz8 = {0, 0, 0, 0, 0, 0, 0, 0};
  __builtin_amdgcn_tensor_load_to_lds(g0, g1, gz4, gz4, gz8, 0);
}

// ---------------------------------------------------------------------------
// Generic GEMM: out[M,N] = act(A[M,K] @ Bw[N,K]^T + bias) (+ residual)
// Block = 256 threads (8 waves), block tile 128x128, wave tile 32x64.
// 8 WMMA per 12 fragment loads per 32-K step.
// ---------------------------------------------------------------------------
template <int ACT>  // 0 = none, 1 = exact GELU
__global__ __launch_bounds__(256) void gemm_wmma_kernel(
    const bf16_t* __restrict__ A, const bf16_t* __restrict__ Bw,
    const float* __restrict__ bias, const float* __restrict__ residual,
    float* __restrict__ outF, bf16_t* __restrict__ outB, int N, int K) {
  int w    = threadIdx.x >> 5;
  int lane = threadIdx.x & 31;
  int wm = blockIdx.x * 128 + (w >> 1) * 32;
  int wn = blockIdx.y * 128 + (w & 1)  * 64;

  v8f c00 = {}, c01 = {}, c02 = {}, c03 = {};
  v8f c10 = {}, c11 = {}, c12 = {}, c13 = {};
  for (int k = 0; k < K; k += 32) {
    v16bf a0 = load_frag_rm(A,  wm,      K, k);
    v16bf a1 = load_frag_rm(A,  wm + 16, K, k);
    v16bf b0 = load_frag_rm(Bw, wn,      K, k);
    v16bf b1 = load_frag_rm(Bw, wn + 16, K, k);
    v16bf b2 = load_frag_rm(Bw, wn + 32, K, k);
    v16bf b3 = load_frag_rm(Bw, wn + 48, K, k);
    c00 = wmma_bf16(a0, b0, c00);
    c01 = wmma_bf16(a0, b1, c01);
    c02 = wmma_bf16(a0, b2, c02);
    c03 = wmma_bf16(a0, b3, c03);
    c10 = wmma_bf16(a1, b0, c10);
    c11 = wmma_bf16(a1, b1, c11);
    c12 = wmma_bf16(a1, b2, c12);
    c13 = wmma_bf16(a1, b3, c13);
  }

  int nloc = lane & 15;
  int rb   = (lane >> 4) << 3;  // rows 0..7 or 8..15 within a 16x16 D tile
#define GEMM_EPILOG(CV, TI, TJ)                                               \
  {                                                                           \
    int col = wn + (TJ) * 16 + nloc;                                          \
    float bi = bias ? bias[col] : 0.0f;                                       \
    _Pragma("unroll") for (int v = 0; v < 8; ++v) {                           \
      int row = wm + (TI) * 16 + rb + v;                                      \
      size_t idx = (size_t)row * (size_t)N + col;                             \
      float val = (CV)[v] + bi;                                               \
      if (ACT == 1) val = gelu_exact(val);                                    \
      if (residual) val += residual[idx];                                     \
      if (outF) outF[idx] = val;                                              \
      if (outB) outB[idx] = (bf16_t)val;                                      \
    }                                                                         \
  }
  GEMM_EPILOG(c00, 0, 0)
  GEMM_EPILOG(c01, 0, 1)
  GEMM_EPILOG(c02, 0, 2)
  GEMM_EPILOG(c03, 0, 3)
  GEMM_EPILOG(c10, 1, 0)
  GEMM_EPILOG(c11, 1, 1)
  GEMM_EPILOG(c12, 1, 2)
  GEMM_EPILOG(c13, 1, 3)
#undef GEMM_EPILOG
}

// ---------------------------------------------------------------------------
// Flash self-attention. q: (BN, C) bf16 (head at col h*64), kv: (BN, 2C) bf16
// (k at h*64, v at 512+h*64). One block = 1 (b, h, 128-q-row chunk); 8 waves,
// 16 q rows per wave; stream over KV in 32-row tiles. K tile loaded by the
// Tensor Data Mover (TENSORcnt); V tile by cooperative transposing load.
// ---------------------------------------------------------------------------
__global__ __launch_bounds__(256) void self_attn_fa_kernel(
    const bf16_t* __restrict__ qb, const bf16_t* __restrict__ kvb,
    float* __restrict__ yF, bf16_t* __restrict__ yB) {
  __shared__ bf16_t Ksh[32 * 64];      // KV-tile rows x d (row major)
  __shared__ bf16_t Vtsh[64 * 32];     // transposed: d x KV-tile rows
  __shared__ float  Psh[8 * 16 * 32];  // per-wave D->A layout staging

  int blk = blockIdx.x;
  int b  = blk >> 6;
  int h  = (blk >> 3) & 7;
  int qc = blk & 7;
  int w    = threadIdx.x >> 5;
  int lane = threadIdx.x & 31;
  int coll = lane & 15;
  int rb   = (lane >> 4) << 3;
  int qrow0 = qc * 128 + w * 16;

  const bf16_t* qbase = qb  + (size_t)b * SEQ * C_DIM + h * HD;        // ld 512
  const bf16_t* kbase = kvb + (size_t)b * SEQ * (2 * C_DIM) + h * HD;  // ld 1024
  const bf16_t* vbase = kbase + C_DIM;

  v16bf q0 = load_frag_rm(qbase, qrow0, C_DIM, 0);
  v16bf q1 = load_frag_rm(qbase, qrow0, C_DIM, 32);

  v8f o0 = {}, o1 = {}, o2 = {}, o3 = {};
  float runM[8], runS[8];
#pragma unroll
  for (int v = 0; v < 8; ++v) { runM[v] = -1e30f; runS[v] = 0.0f; }

  float* pw = &Psh[w * 512];

  for (int nt = 0; nt < SEQ; nt += 32) {
    // --- K tile via TDM (issued once, by wave 0) ---
    if (threadIdx.x < 32)
      tdm_load_k_tile(kbase + (size_t)nt * (2 * C_DIM), Ksh);

    // --- V tile: cooperative transposing load (TDM cannot transpose) ---
    {
      int r  = threadIdx.x >> 3;        // 0..31
      int cc = (threadIdx.x & 7) * 8;   // 0..56
      v8bf vv8 = *(const v8bf*)(vbase + (size_t)(nt + r) * (2 * C_DIM) + cc);
#pragma unroll
      for (int i = 0; i < 8; ++i) Vtsh[(cc + i) * 32 + r] = vv8[i];
      if (nt + 32 < SEQ)  // global_prefetch_b8 for next tile
        __builtin_prefetch(vbase + (size_t)(nt + 32 + r) * (2 * C_DIM) + cc, 0, 1);
    }
    if (threadIdx.x < 32) __builtin_amdgcn_s_wait_tensorcnt(0);
    __syncthreads();

    // --- scores S = q @ K^T (K-dim = d = 64) ---
    v8f s0 = {}, s1 = {};
    {
      v16bf kf;
      kf = load_frag_rm(Ksh, 0,  64, 0);  s0 = wmma_bf16(q0, kf, s0);
      kf = load_frag_rm(Ksh, 0,  64, 32); s0 = wmma_bf16(q1, kf, s0);
      kf = load_frag_rm(Ksh, 16, 64, 0);  s1 = wmma_bf16(q0, kf, s1);
      kf = load_frag_rm(Ksh, 16, 64, 32); s1 = wmma_bf16(q1, kf, s1);
    }

    // --- online softmax over this 32-column chunk ---
#pragma unroll
    for (int v = 0; v < 8; ++v) {
      float a  = s0[v] * 0.125f;  // d^-0.5 = 1/8
      float c2 = s1[v] * 0.125f;
      float cm = fmaxf(a, c2);
#pragma unroll
      for (int off = 1; off < 16; off <<= 1)
        cm = fmaxf(cm, __shfl_xor(cm, off, 32));
      float nM = fmaxf(runM[v], cm);
      float sc = __expf(runM[v] - nM);
      float p0 = __expf(a - nM);
      float p1 = __expf(c2 - nM);
      float cs = p0 + p1;
#pragma unroll
      for (int off = 1; off < 16; off <<= 1) cs += __shfl_xor(cs, off, 32);
      runS[v] = runS[v] * sc + cs;
      runM[v] = nM;
      o0[v] *= sc; o1[v] *= sc; o2[v] *= sc; o3[v] *= sc;
      int rowl = v + rb;
      pw[rowl * 32 + coll]      = p0;
      pw[rowl * 32 + 16 + coll] = p1;
    }

    // --- convert P from D-layout (in LDS) to A-layout bf16 fragment ---
    v16bf pf;
    {
      const float* pp = &pw[(lane & 15) * 32 + rb];
#pragma unroll
      for (int i = 0; i < 8; ++i) {
        pf[i]     = (bf16_t)pp[i];
        pf[8 + i] = (bf16_t)pp[16 + i];
      }
    }

    // --- O += P @ V (K-dim = 32 kv rows) ---
    {
      v16bf vf;
      vf = load_frag_rm(Vtsh, 0,  32, 0); o0 = wmma_bf16(pf, vf, o0);
      vf = load_frag_rm(Vtsh, 16, 32, 0); o1 = wmma_bf16(pf, vf, o1);
      vf = load_frag_rm(Vtsh, 32, 32, 0); o2 = wmma_bf16(pf, vf, o2);
      vf = load_frag_rm(Vtsh, 48, 32, 0); o3 = wmma_bf16(pf, vf, o3);
    }
    __syncthreads();
  }

  // --- normalize & store ---
#pragma unroll
  for (int v = 0; v < 8; ++v) {
    int row = qrow0 + rb + v;
    float inv = 1.0f / runS[v];
    size_t base = ((size_t)b * SEQ + row) * C_DIM + h * HD + coll;
    float f0 = o0[v] * inv, f1 = o1[v] * inv, f2 = o2[v] * inv, f3 = o3[v] * inv;
    yF[base]      = f0; yF[base + 16] = f1; yF[base + 32] = f2; yF[base + 48] = f3;
    yB[base]      = (bf16_t)f0; yB[base + 16] = (bf16_t)f1;
    yB[base + 32] = (bf16_t)f2; yB[base + 48] = (bf16_t)f3;
  }
}

// ---------------------------------------------------------------------------
// Elementwise / small kernels
// ---------------------------------------------------------------------------
__global__ __launch_bounds__(256) void k_cvt_f32_bf16(
    const float* __restrict__ in, bf16_t* __restrict__ out, size_t n) {
  size_t i = (size_t)blockIdx.x * 256 + threadIdx.x;
  if (i < n) out[i] = (bf16_t)in[i];
}

// cat[r, 0:512] = ys[r], cat[r, 512:1024] = yo[r]  (bf16)
__global__ __launch_bounds__(256) void k_concat(
    const float* __restrict__ ys, const float* __restrict__ yo,
    bf16_t* __restrict__ cat) {
  size_t i = (size_t)blockIdx.x * 256 + threadIdx.x;  // over BN*512
  size_t r = i >> 9;
  int    c = (int)(i & 511);
  cat[r * 1024 + c]       = (bf16_t)ys[i];
  cat[r * 1024 + 512 + c] = (bf16_t)yo[i];
}

// row-softmax over 512 cols, in-place
__global__ __launch_bounds__(256) void k_softmax_rows512(float* __restrict__ x) {
  __shared__ float red[256];
  int tid = threadIdx.x;
  float* row = x + (size_t)blockIdx.x * 512;
  float a = row[tid], b = row[tid + 256];
  red[tid] = fmaxf(a, b);
  __syncthreads();
  for (int s = 128; s > 0; s >>= 1) {
    if (tid < s) red[tid] = fmaxf(red[tid], red[tid + s]);
    __syncthreads();
  }
  float m = red[0];
  __syncthreads();
  float e0 = __expf(a - m), e1 = __expf(b - m);
  red[tid] = e0 + e1;
  __syncthreads();
  for (int s = 128; s > 0; s >>= 1) {
    if (tid < s) red[tid] += red[tid + s];
    __syncthreads();
  }
  float inv = 1.0f / red[0];
  row[tid] = e0 * inv;
  row[tid + 256] = e1 * inv;
}

// k_in[r,0]=kn+ys, k_in[r,1]=ys*rel ; v_in[r,0]=vn+ys, v_in[r,1]=yo  (bf16)
__global__ __launch_bounds__(256) void k_build_kv_in(
    const float* __restrict__ ys, const float* __restrict__ yo,
    const float* __restrict__ rel, const float* __restrict__ knrow,
    const float* __restrict__ vnrow, bf16_t* __restrict__ kin,
    bf16_t* __restrict__ vin) {
  size_t i = (size_t)blockIdx.x * 256 + threadIdx.x;  // over BN*512
  size_t r = i >> 9;
  int    c = (int)(i & 511);
  float y = ys[i];
  kin[(r * 2) * 512 + c]     = (bf16_t)(knrow[c] + y);
  kin[(r * 2 + 1) * 512 + c] = (bf16_t)(y * rel[i]);
  vin[(r * 2) * 512 + c]     = (bf16_t)(vnrow[c] + y);
  vin[(r * 2 + 1) * 512 + c] = (bf16_t)yo[i];
}

// K=2 cross attention combine: one thread per (b, n, h)
__global__ __launch_bounds__(256) void k_cross_combine(
    const float* __restrict__ qp, const float* __restrict__ kp,
    const float* __restrict__ vp, bf16_t* __restrict__ ob) {
  int gid = blockIdx.x * 256 + threadIdx.x;  // BN*HEADS = 65536
  size_t r = (size_t)(gid >> 3);
  int h = gid & 7;
  const float* q  = qp + r * 512 + h * HD;
  const float* k0 = kp + (r * 2) * 512 + h * HD;
  const float* k1 = k0 + 512;
  float d0 = 0.0f, d1 = 0.0f;
#pragma unroll 8
  for (int d = 0; d < HD; ++d) {
    float qd = q[d];
    d0 += qd * k0[d];
    d1 += qd * k1[d];
  }
  d0 *= 0.125f; d1 *= 0.125f;
  float m = fmaxf(d0, d1);
  float e0 = __expf(d0 - m), e1 = __expf(d1 - m);
  float inv = 1.0f / (e0 + e1);
  float w0 = e0 * inv, w1 = e1 * inv;
  const float* v0 = vp + (r * 2) * 512 + h * HD;
  const float* v1 = v0 + 512;
  bf16_t* o = ob + r * 512 + h * HD;
#pragma unroll 8
  for (int d = 0; d < HD; ++d) o[d] = (bf16_t)(w0 * v0[d] + w1 * v1[d]);
}

// ---------------------------------------------------------------------------
// Host orchestration
// ---------------------------------------------------------------------------
extern "C" void kernel_launch(void* const* d_in, const int* in_sizes, int n_in,
                              void* d_out, int out_size, void* d_ws,
                              size_t ws_size, hipStream_t stream) {
  (void)in_sizes; (void)n_in; (void)out_size; (void)ws_size;
  const float* x0 = (const float*)d_in[0];
  const float* x1 = (const float*)d_in[1];
  const float* Wq  = (const float*)d_in[4];
  const float* Wkv = (const float*)d_in[5];
  const float* Wj1 = (const float*)d_in[6];
  const float* bj1 = (const float*)d_in[7];
  const float* Wj2 = (const float*)d_in[8];
  const float* bj2 = (const float*)d_in[9];
  const float* kn  = (const float*)d_in[10];
  const float* vn  = (const float*)d_in[11];
  const float* Wcin[2]  = {(const float*)d_in[12], (const float*)d_in[16]};
  const float* bcin[2]  = {(const float*)d_in[13], (const float*)d_in[17]};
  const float* Wcout[2] = {(const float*)d_in[14], (const float*)d_in[18]};
  const float* bcout[2] = {(const float*)d_in[15], (const float*)d_in[19]};
  const float* Wp = (const float*)d_in[20];
  const float* bp = (const float*)d_in[21];

  char* ws = (char*)d_ws;
  size_t off = 0;
  auto alloc = [&](size_t bytes) -> void* {
    void* p = ws + off;
    off += (bytes + 255) & ~(size_t)255;
    return p;
  };
  const size_t S1 = (size_t)BN * C_DIM * sizeof(bf16_t);
  const size_t S2 = (size_t)BN * 2 * C_DIM * sizeof(bf16_t);
  const size_t F1 = (size_t)BN * C_DIM * sizeof(float);
  const size_t F2 = (size_t)BN * 2 * C_DIM * sizeof(float);

  bf16_t* wqB  = (bf16_t*)alloc((size_t)C_DIM * C_DIM * 2);
  bf16_t* wkvB = (bf16_t*)alloc((size_t)2 * C_DIM * C_DIM * 2);
  bf16_t* wj1B = (bf16_t*)alloc((size_t)C_DIM * 2 * C_DIM * 2);
  bf16_t* wj2B = (bf16_t*)alloc((size_t)C_DIM * C_DIM * 2);
  bf16_t* wcinB[2], *wcoutB[2];
  wcinB[0]  = (bf16_t*)alloc((size_t)3 * C_DIM * C_DIM * 2);
  wcoutB[0] = (bf16_t*)alloc((size_t)C_DIM * C_DIM * 2);
  wcinB[1]  = (bf16_t*)alloc((size_t)3 * C_DIM * C_DIM * 2);
  wcoutB[1] = (bf16_t*)alloc((size_t)C_DIM * C_DIM * 2);
  bf16_t* wpB = (bf16_t*)alloc((size_t)C_DIM * C_DIM * 2);

  bf16_t* xb   = (bf16_t*)alloc(S1);
  bf16_t* qb   = (bf16_t*)alloc(S1);
  bf16_t* kvb  = (bf16_t*)alloc(S2);
  float*  yF[2] = {(float*)alloc(F1), (float*)alloc(F1)};
  bf16_t* yB[2] = {(bf16_t*)alloc(S1), (bf16_t*)alloc(S1)};
  float*  rel[2] = {(float*)alloc(F1), (float*)alloc(F1)};
  bf16_t* catb = (bf16_t*)alloc(S2);
  bf16_t* hdnb = (bf16_t*)alloc(S1);
  float*  qpf  = (float*)alloc(F1);
  bf16_t* kinb = (bf16_t*)alloc(S2);
  float*  kpf  = (float*)alloc(F2);
  bf16_t* vinb = (bf16_t*)alloc(S2);
  float*  vpf  = (float*)alloc(F2);
  bf16_t* ob   = (bf16_t*)alloc(S1);
  bf16_t* newb = (bf16_t*)alloc(S1);

  dim3 blk(256);
  auto cvt = [&](const float* src, bf16_t* dst, size_t n) {
    k_cvt_f32_bf16<<<dim3((unsigned)((n + 255) / 256)), blk, 0, stream>>>(src, dst, n);
  };
  auto gemm = [&](int act, const bf16_t* A, const bf16_t* Bw, const float* bias,
                  const float* resid, float* oF, bf16_t* oB, int M, int N, int K) {
    dim3 g(M / 128, N / 128);
    if (act == 1)
      gemm_wmma_kernel<1><<<g, blk, 0, stream>>>(A, Bw, bias, resid, oF, oB, N, K);
    else
      gemm_wmma_kernel<0><<<g, blk, 0, stream>>>(A, Bw, bias, resid, oF, oB, N, K);
  };

  // --- weights to bf16 ---
  cvt(Wq, wqB, (size_t)512 * 512);
  cvt(Wkv, wkvB, (size_t)1024 * 512);
  cvt(Wj1, wj1B, (size_t)512 * 1024);
  cvt(Wj2, wj2B, (size_t)512 * 512);
  cvt(Wcin[0], wcinB[0], (size_t)1536 * 512);
  cvt(Wcout[0], wcoutB[0], (size_t)512 * 512);
  cvt(Wcin[1], wcinB[1], (size_t)1536 * 512);
  cvt(Wcout[1], wcoutB[1], (size_t)512 * 512);
  cvt(Wp, wpB, (size_t)512 * 512);

  // --- phase 1: self attention per stream ---
  const float* xin[2] = {x0, x1};
  for (int s = 0; s < 2; ++s) {
    cvt(xin[s], xb, (size_t)BN * 512);
    gemm(0, xb, wqB, nullptr, nullptr, nullptr, qb, BN, 512, 512);
    gemm(0, xb, wkvB, nullptr, nullptr, nullptr, kvb, BN, 1024, 512);
    self_attn_fa_kernel<<<dim3(512), blk, 0, stream>>>(qb, kvb, yF[s], yB[s]);
  }

  // --- phase 2: judger -> rel (row softmax over C) ---
  unsigned ew_blocks = (unsigned)((size_t)BN * 512 / 256);
  for (int s = 0; s < 2; ++s) {
    int o = 1 - s;
    k_concat<<<dim3(ew_blocks), blk, 0, stream>>>(yF[s], yF[o], catb);
    gemm(1, catb, wj1B, bj1, nullptr, nullptr, hdnb, BN, 512, 1024);
    gemm(0, hdnb, wj2B, bj2, nullptr, rel[s], nullptr, BN, 512, 512);
    k_softmax_rows512<<<dim3(BN), blk, 0, stream>>>(rel[s]);
  }

  // --- phase 3: cross attention + residual + final projection ---
  for (int s = 0; s < 2; ++s) {
    int o = 1 - s;
    gemm(0, yB[s], wcinB[s], bcin[s], nullptr, qpf, nullptr, BN, 512, 512);
    k_build_kv_in<<<dim3(ew_blocks), blk, 0, stream>>>(
        yF[s], yF[o], rel[s], kn + s * 512, vn + s * 512, kinb, vinb);
    gemm(0, kinb, wcinB[s] + (size_t)512 * 512, bcin[s] + 512, nullptr, kpf,
         nullptr, 2 * BN, 512, 512);
    gemm(0, vinb, wcinB[s] + (size_t)2 * 512 * 512, bcin[s] + 1024, nullptr,
         vpf, nullptr, 2 * BN, 512, 512);
    k_cross_combine<<<dim3(BN * HEADS / 256), blk, 0, stream>>>(qpf, kpf, vpf, ob);
    // new = y + (o @ Wout^T + bout), stored as bf16
    gemm(0, ob, wcoutB[s], bcout[s], yF[s], nullptr, newb, BN, 512, 512);
    // out_s = new @ Wp^T + bp  (f32 into d_out)
    gemm(0, newb, wpB, bp, nullptr, (float*)d_out + (size_t)s * BN * 512,
         nullptr, BN, 512, 512);
  }
}
